// Encoder_4063039062793
// MI455X (gfx1250) — matrix operand order repaired and checked by reading, not compiled, not run on hardware
//
#include <hip/hip_runtime.h>
#include <cstddef>
#include <cstdint>

// ---------------------------------------------------------------------------
// MI455X (gfx1250): dense matmuls via v_wmma_f32_16x16x32_f16 (f16 operands
// staged in LDS, f32 accumulate); wave32 VALU kernels for frames/KNN/pair/attn.
// ---------------------------------------------------------------------------

typedef __attribute__((ext_vector_type(16))) _Float16 v16h;
typedef __attribute__((ext_vector_type(8)))  _Float16 v8h;
typedef __attribute__((ext_vector_type(8)))  float    v8f;

constexpr int NRES = 8192;
constexpr int KNB  = 32;
constexpr int LCH  = 256;
constexpr int PCH  = 64;
constexpr int HEADS = 8;
constexpr int NK   = NRES * KNB;      // 262144 pair rows
constexpr float FEPS = 1e-6f;
constexpr int LS = 40;                // LDS tile row stride (halfs), conflict pad

#define FG_GELU 1
#define FG_BIAS 2
#define FG_RES  4

// Async global->LDS: builtin confirmed present on this toolchain; params are
// (global int* src, local int* dst, imm offset, imm cpol).
#if defined(__HIP_DEVICE_COMPILE__) && defined(__has_builtin)
#if __has_builtin(__builtin_amdgcn_global_load_async_to_lds_b32) && \
    __has_builtin(__builtin_amdgcn_s_wait_asynccnt)
#define ASYNC_LDS 1
#endif
#endif
#ifndef ASYNC_LDS
#define ASYNC_LDS 0
#endif

#if ASYNC_LDS
typedef __attribute__((address_space(1))) int as1_int;
typedef __attribute__((address_space(3))) int as3_int;
#endif

__device__ __forceinline__ float gelu_f(float x) {
  float x3 = x * x * x;
  return 0.5f * x * (1.f + tanhf(0.7978845608028654f * (x + 0.044715f * x3)));
}

// ---------------------------------------------------------------------------
// WMMA GEMM: C[M x Nn] = epilogue(A[M x Kd] @ B[Kd x Nn]); row-major f32 I/O.
// Block: 256 threads = 8 waves; tile 64x64; wave (wm,wn) owns rows wm*16..+15
// and two 16-wide column tiles at wn*32 and wn*32+16.
// Fast path: branch-free float4 (b128) staging; slow path: clamped + select.
// ---------------------------------------------------------------------------
__global__ __launch_bounds__(256) void wmma_gemm_kernel(
    const float* __restrict__ A, int lda,
    const float* __restrict__ B, int ldb,
    const float* __restrict__ bias,
    const float* __restrict__ res,
    float* __restrict__ C, int ldc,
    int M, int Kd, int Nn, int flags)
{
  __shared__ __align__(16) _Float16 As[64 * LS];
  __shared__ __align__(16) _Float16 Bs[64 * LS];

  const int tid  = threadIdx.x;
  const int lane = tid & 31;
  const int w    = tid >> 5;
  const int wm   = w >> 1;   // 0..3  (16-row group)
  const int wn   = w & 1;    // 0..1  (32-col group)
  const int m0   = blockIdx.x * 64;
  const int n0   = blockIdx.y * 64;

  // A staging: thread -> (row alr, 8 consecutive k at alk)
  const int alr = tid >> 2;         // 0..63
  const int alk = (tid & 3) << 3;   // 0,8,16,24
  // B staging: thread -> (k = bkk, 8 consecutive n at bnn); stored transposed.
  const int bkk = tid & 31;         // 0..31
  const int bnn = (tid >> 5) << 3;  // 0,8,..,56

  const int arow = m0 + alr;
  const bool fullMN = (m0 + 64 <= M) && (n0 + 64 <= Nn) &&
                      ((lda & 3) == 0) && ((ldb & 3) == 0);

  v8f c0 = {0.f,0.f,0.f,0.f,0.f,0.f,0.f,0.f};
  v8f c1 = {0.f,0.f,0.f,0.f,0.f,0.f,0.f,0.f};

  for (int k0 = 0; k0 < Kd; k0 += 32) {
    if (fullMN && (k0 + 32 <= Kd)) {
      // ---- fast path: wide vector loads, no predication ----
      const float4* ap = (const float4*)&A[(size_t)arow * lda + k0 + alk];
      float4 a0 = ap[0], a1 = ap[1];
      v8h ah;
      ah[0]=(_Float16)a0.x; ah[1]=(_Float16)a0.y; ah[2]=(_Float16)a0.z; ah[3]=(_Float16)a0.w;
      ah[4]=(_Float16)a1.x; ah[5]=(_Float16)a1.y; ah[6]=(_Float16)a1.z; ah[7]=(_Float16)a1.w;
      *(v8h*)&As[alr * LS + alk] = ah;

      const float4* bp = (const float4*)&B[(size_t)(k0 + bkk) * ldb + n0 + bnn];
      float4 b0 = bp[0], b1 = bp[1];
      Bs[(bnn+0)*LS + bkk] = (_Float16)b0.x;
      Bs[(bnn+1)*LS + bkk] = (_Float16)b0.y;
      Bs[(bnn+2)*LS + bkk] = (_Float16)b0.z;
      Bs[(bnn+3)*LS + bkk] = (_Float16)b0.w;
      Bs[(bnn+4)*LS + bkk] = (_Float16)b1.x;
      Bs[(bnn+5)*LS + bkk] = (_Float16)b1.y;
      Bs[(bnn+6)*LS + bkk] = (_Float16)b1.z;
      Bs[(bnn+7)*LS + bkk] = (_Float16)b1.w;

      if (k0 + 64 <= Kd) {
        __builtin_prefetch(&A[(size_t)arow * lda + k0 + 32 + alk], 0, 0);
        __builtin_prefetch(&B[(size_t)(k0 + 32 + bkk) * ldb + n0 + bnn], 0, 0);
      }
    } else {
      // ---- slow path: clamp address in-bounds, select-zero (no branches) ----
      const int sr = arow < M ? arow : (M - 1);
      #pragma unroll
      for (int j = 0; j < 8; ++j) {
        int gk = k0 + alk + j;
        int sk = gk < Kd ? gk : (Kd - 1);
        float av = A[(size_t)sr * lda + sk];
        As[alr * LS + alk + j] =
            (_Float16)((arow < M && gk < Kd) ? av : 0.f);
      }
      const int gkb = k0 + bkk;
      const int skb = gkb < Kd ? gkb : (Kd - 1);
      #pragma unroll
      for (int j = 0; j < 8; ++j) {
        int gn = n0 + bnn + j;
        int sn = gn < Nn ? gn : (Nn - 1);
        float bv = B[(size_t)skb * ldb + sn];
        Bs[(bnn + j) * LS + bkk] =
            (_Float16)((gkb < Kd && gn < Nn) ? bv : 0.f);
      }
    }
    __syncthreads();

    const int rc = lane & 15;
    const int hi = lane >> 4;     // lane half

    // A fragment: row = wm*16+rc; K elems {hi*8..+7} and {16+hi*8..+7}
    v16h afrag;
    {
      const _Float16* apf = &As[(wm * 16 + rc) * LS + hi * 8];
      v8h alo = *(const v8h*)apf;
      v8h ahi2 = *(const v8h*)(apf + 16);
      #pragma unroll
      for (int j = 0; j < 8; ++j) { afrag[j] = alo[j]; afrag[8 + j] = ahi2[j]; }
    }
    // B fragments: col = base+rc; K elems {hi*8..+7} and {16+hi*8..+7}
    v16h b0f, b1f;
    {
      const _Float16* bpf = &Bs[(wn * 32 + rc) * LS + hi * 8];
      v8h bl = *(const v8h*)bpf;
      v8h bh = *(const v8h*)(bpf + 16);
      #pragma unroll
      for (int j = 0; j < 8; ++j) { b0f[j] = bl[j]; b0f[8 + j] = bh[j]; }
      const _Float16* bpf2 = &Bs[(wn * 32 + 16 + rc) * LS + hi * 8];
      v8h bl2 = *(const v8h*)bpf2;
      v8h bh2 = *(const v8h*)(bpf2 + 16);
      #pragma unroll
      for (int j = 0; j < 8; ++j) { b1f[j] = bl2[j]; b1f[8 + j] = bh2[j]; }
    }

    c0 = __builtin_amdgcn_wmma_f32_16x16x32_f16(false, afrag, false, b0f,
                                                (short)0, c0, false, false);
    c1 = __builtin_amdgcn_wmma_f32_16x16x32_f16(false, afrag, false, b1f,
                                                (short)0, c1, false, false);
    __syncthreads();
  }

  // Epilogue: lane holds col = lane&15, rows (lane>>4)*8 + r in its 8 VGPRs.
  const int rc  = lane & 15;
  const int rb  = (lane >> 4) * 8;
  const int gn0 = n0 + wn * 32 + rc;
  const int gn1 = gn0 + 16;
  #pragma unroll
  for (int r = 0; r < 8; ++r) {
    int gm = m0 + wm * 16 + rb + r;
    if (gm >= M) continue;
    if (gn0 < Nn) {
      float v = c0[r];
      if (flags & FG_BIAS) v += bias[gn0];
      if (flags & FG_GELU) v = gelu_f(v);
      if (flags & FG_RES)  v += res[(size_t)gm * ldc + gn0];
      C[(size_t)gm * ldc + gn0] = v;
    }
    if (gn1 < Nn) {
      float v = c1[r];
      if (flags & FG_BIAS) v += bias[gn1];
      if (flags & FG_GELU) v = gelu_f(v);
      if (flags & FG_RES)  v += res[(size_t)gm * ldc + gn1];
      C[(size_t)gm * ldc + gn1] = v;
    }
  }
}

// ---------------------------------------------------------------------------
// Per-residue frames + augmented coords + pos21 + 443-dim features.
// ---------------------------------------------------------------------------
__global__ __launch_bounds__(256) void prep_kernel(
    const float* __restrict__ pos, const unsigned char* __restrict__ is_target,
    const float* __restrict__ hotspots, const int* __restrict__ aa_gt,
    const float* __restrict__ vec_w,
    float* __restrict__ Rb, float* __restrict__ Tb,
    float* __restrict__ loc2, float* __restrict__ pos21,
    float* __restrict__ feats)
{
  int n = blockIdx.x * blockDim.x + threadIdx.x;
  if (n >= NRES) return;

  float p[14][3];
  #pragma unroll
  for (int a = 0; a < 14; ++a)
    for (int d = 0; d < 3; ++d) p[a][d] = pos[(size_t)n * 42 + a * 3 + d];

  float t[3] = { p[1][0], p[1][1], p[1][2] };
  float e1[3], e2[3], e3[3], v2[3];
  for (int d = 0; d < 3; ++d) e1[d] = p[2][d] - t[d];
  float n1 = sqrtf(e1[0]*e1[0]+e1[1]*e1[1]+e1[2]*e1[2] + 1e-12f);
  for (int d = 0; d < 3; ++d) e1[d] /= (n1 + FEPS);
  for (int d = 0; d < 3; ++d) v2[d] = p[0][d] - t[d];
  float dp = v2[0]*e1[0]+v2[1]*e1[1]+v2[2]*e1[2];
  for (int d = 0; d < 3; ++d) v2[d] -= dp * e1[d];
  float n2 = sqrtf(v2[0]*v2[0]+v2[1]*v2[1]+v2[2]*v2[2] + 1e-12f);
  for (int d = 0; d < 3; ++d) e2[d] = v2[d] / (n2 + FEPS);
  e3[0] = e1[1]*e2[2] - e1[2]*e2[1];
  e3[1] = e1[2]*e2[0] - e1[0]*e2[2];
  e3[2] = e1[0]*e2[1] - e1[1]*e2[0];
  float Rm[3][3];
  for (int j = 0; j < 3; ++j) { Rm[j][0] = e1[j]; Rm[j][1] = e2[j]; Rm[j][2] = e3[j]; }

  float loc0[14][3];
  #pragma unroll
  for (int a = 0; a < 14; ++a)
    for (int i = 0; i < 3; ++i) {
      float s = 0.f;
      for (int j = 0; j < 3; ++j) s += (p[a][j] - t[j]) * Rm[j][i];
      loc0[a][i] = s;
    }

  float aug[16][3]; float ms = 0.f;
  #pragma unroll
  for (int o = 0; o < 16; ++o)
    for (int d = 0; d < 3; ++d) {
      float s = 0.f;
      for (int a = 0; a < 14; ++a) s += loc0[a][d] * vec_w[a * 16 + o];
      aug[o][d] = s;
    }
  for (int o = 0; o < 16; ++o)
    ms += aug[o][0]*aug[o][0] + aug[o][1]*aug[o][1] + aug[o][2]*aug[o][2];
  ms = sqrtf(ms * (1.f/16.f) + FEPS);
  for (int o = 0; o < 16; ++o)
    for (int d = 0; d < 3; ++d) aug[o][d] /= ms;

  float loc[21][3];
  for (int a = 0; a < 5; ++a) for (int d = 0; d < 3; ++d) loc[a][d] = loc0[a][d];
  for (int o = 0; o < 16; ++o) for (int d = 0; d < 3; ++d) loc[5+o][d] = aug[o][d];

  for (int a = 0; a < 21; ++a)
    for (int j = 0; j < 3; ++j)
      pos21[(size_t)n*63 + a*3 + j] =
        Rm[j][0]*loc[a][0] + Rm[j][1]*loc[a][1] + Rm[j][2]*loc[a][2] + t[j];

  for (int j = 0; j < 3; ++j) {
    Tb[(size_t)n*3 + j] = t[j];
    for (int i = 0; i < 3; ++i) Rb[(size_t)n*9 + j*3 + i] = Rm[j][i];
  }
  for (int a = 0; a < 21; ++a)
    for (int d = 0; d < 3; ++d) loc2[(size_t)n*63 + a*3 + d] = loc[a][d];

  float dist[21];
  for (int a = 0; a < 21; ++a)
    dist[a] = sqrtf(loc[a][0]*loc[a][0]+loc[a][1]*loc[a][1]+loc[a][2]*loc[a][2] + 1e-12f);

  float* f = feats + (size_t)n * 448;
  for (int a = 0; a < 21; ++a)
    for (int d = 0; d < 3; ++d) f[a*3+d] = loc[a][d] / (dist[a] + FEPS);
  for (int a = 0; a < 21; ++a)
    for (int c = 0; c < 16; ++c) {
      float z = (dist[a] - 22.f * c * (1.f/15.f)) * (1.f/1.375f);
      f[63 + a*16 + c] = __expf(-z*z);
    }
  for (int a = 0; a < 21; ++a) f[399 + a] = logf(dist[a] + 1.f);
  int it = is_target[n] ? 1 : 0;
  f[420] = (float)it;
  f[421] = hotspots[n];
  int aav = it ? aa_gt[n] : 20;
  for (int c = 0; c < 21; ++c) f[422 + c] = (c == aav) ? 1.f : 0.f;
  for (int c = 443; c < 448; ++c) f[c] = 0.f;
}

// ---------------------------------------------------------------------------
// KNN: one workgroup per query; LDS distance array + 32x packed-key argmin.
// ---------------------------------------------------------------------------
__global__ __launch_bounds__(256) void knn_kernel(
    const float* __restrict__ pos21, const int* __restrict__ batch,
    const unsigned char* __restrict__ is_target,
    const unsigned char* __restrict__ mask, int* __restrict__ nb)
{
  __shared__ float sdist[NRES];
  __shared__ unsigned long long red[256];
  const int n = blockIdx.x, tid = threadIdx.x;
  const float qx = pos21[(size_t)n*63 + 12];
  const float qy = pos21[(size_t)n*63 + 13];
  const float qz = pos21[(size_t)n*63 + 14];
  const int bq = batch[n];
  const int tq = is_target[n] ? 1 : 0;
  const int mq = mask[n] ? 1 : 0;

  for (int j = tid; j < NRES; j += 256) {
    float dx = pos21[(size_t)j*63 + 12] - qx;
    float dy = pos21[(size_t)j*63 + 13] - qy;
    float dz = pos21[(size_t)j*63 + 14] - qz;
    float d  = sqrtf(dx*dx + dy*dy + dz*dz + 1e-12f);
    bool bad = (batch[j] != bq) || (tq && !is_target[j]) || !(mq && mask[j]);
    sdist[j] = bad ? __int_as_float(0x7f800000) : d;
  }
  __syncthreads();

  for (int it = 0; it < KNB; ++it) {
    unsigned long long best = ~0ull;
    for (int j = tid; j < NRES; j += 256) {
      unsigned long long key =
        ((unsigned long long)__float_as_uint(sdist[j]) << 32) | (unsigned)j;
      best = (key < best) ? key : best;
    }
    red[tid] = best;
    __syncthreads();
    for (int s = 128; s > 0; s >>= 1) {
      if (tid < s) { if (red[tid + s] < red[tid]) red[tid] = red[tid + s]; }
      __syncthreads();
    }
    unsigned long long top = red[0];
    int idx = (int)(top & 0xffffffffu);
    unsigned db = (unsigned)(top >> 32);
    if (tid == 0) {
      nb[(size_t)n * KNB + it] = (db >= 0x7f800000u) ? -1 : idx;
      sdist[idx] = __int_as_float(0x7f800000);
    }
    __syncthreads();
  }
}

// ---------------------------------------------------------------------------
// Pair features: one wave per (n,k) pair row; fused LayerNorm over 64 ch.
// ---------------------------------------------------------------------------
__global__ __launch_bounds__(256) void pair_kernel(
    const float* __restrict__ pos21, const float* __restrict__ Rb,
    const float* __restrict__ Tb, const int* __restrict__ nb,
    const int* __restrict__ resi, const int* __restrict__ chain,
    const unsigned char* __restrict__ mask,
    const float* __restrict__ w_rel, const float* __restrict__ w_dist,
    const float* __restrict__ w_dir, const float* __restrict__ w_vec,
    const float* __restrict__ w_rot,
    const float* __restrict__ pln_s, const float* __restrict__ pln_o,
    float* __restrict__ pair, int* __restrict__ pm)
{
  const int wid  = (blockIdx.x * 256 + threadIdx.x) >> 5;  // pair row
  const int lane = threadIdx.x & 31;
  if (wid >= NK) return;
  const int n  = wid >> 5;
  const int idx = nb[wid];
  const bool valid = (idx != -1);
  const int i = valid ? idx : 0;

  if (lane == 0) pm[wid] = (mask[n] && mask[i] && valid) ? 1 : 0;

  int rel = resi[i] - resi[n];
  rel = (rel < -32 ? -32 : (rel > 32 ? 32 : rel)) + 32;
  const bool use_rel = (chain[i] == chain[n]) && valid;
  float acc0 = use_rel ? w_rel[rel * PCH + lane]      : 0.f;
  float acc1 = use_rel ? w_rel[rel * PCH + lane + 32] : 0.f;

  float pa[5][3], pb[5][3];
  #pragma unroll
  for (int a = 0; a < 5; ++a)
    for (int d = 0; d < 3; ++d) {
      pa[a][d] = pos21[(size_t)n*63 + a*3 + d];
      pb[a][d] = pos21[(size_t)i*63 + a*3 + d];
    }

  // RBF(dist) @ w_dist  (400 -> 64)
  #pragma unroll
  for (int a = 0; a < 5; ++a) {
    #pragma unroll
    for (int b = 0; b < 5; ++b) {
      float dx = pa[a][0]-pb[b][0], dy = pa[a][1]-pb[b][1], dz = pa[a][2]-pb[b][2];
      float d = sqrtf(dx*dx + dy*dy + dz*dz + 1e-12f);
      for (int c = 0; c < 16; ++c) {
        float z = (d - 22.f * c * (1.f/15.f)) * (1.f/1.375f);
        float r = __expf(-z*z);
        int fi = a*80 + b*16 + c;
        acc0 += r * w_dist[fi * PCH + lane];
        acc1 += r * w_dist[fi * PCH + lane + 32];
      }
    }
  }

  float Rm[9], Ri[9], tn[3];
  #pragma unroll
  for (int j = 0; j < 9; ++j) { Rm[j] = Rb[(size_t)n*9 + j]; Ri[j] = Rb[(size_t)i*9 + j]; }
  for (int d = 0; d < 3; ++d) tn[d] = Tb[(size_t)n*3 + d];

  // dirs / vloc projections (15 each)
  #pragma unroll
  for (int b = 0; b < 5; ++b) {
    float v0 = pb[b][0]-tn[0], v1 = pb[b][1]-tn[1], v2 = pb[b][2]-tn[2];
    float vl[3];
    for (int ii = 0; ii < 3; ++ii)
      vl[ii] = v0*Rm[0*3+ii] + v1*Rm[1*3+ii] + v2*Rm[2*3+ii];
    float nrm = sqrtf(vl[0]*vl[0]+vl[1]*vl[1]+vl[2]*vl[2] + 1e-12f) + FEPS;
    for (int ii = 0; ii < 3; ++ii) {
      int fi = b*3 + ii;
      float dir = vl[ii] / nrm;
      acc0 += dir * w_dir[fi*PCH + lane]      + vl[ii] * w_vec[fi*PCH + lane];
      acc1 += dir * w_dir[fi*PCH + lane + 32] + vl[ii] * w_vec[fi*PCH + lane + 32];
    }
  }

  // rrel = R[n]^T R[i]  (9)
  #pragma unroll
  for (int a = 0; a < 3; ++a)
    for (int b = 0; b < 3; ++b) {
      float rr = Rm[0*3+a]*Ri[0*3+b] + Rm[1*3+a]*Ri[1*3+b] + Rm[2*3+a]*Ri[2*3+b];
      int fi = a*3 + b;
      acc0 += rr * w_rot[fi*PCH + lane];
      acc1 += rr * w_rot[fi*PCH + lane + 32];
    }

  // fused LayerNorm over 64 channels (2 per lane)
  float s = acc0 + acc1, sq = acc0*acc0 + acc1*acc1;
  for (int m = 16; m; m >>= 1) { s += __shfl_xor(s, m, 32); sq += __shfl_xor(sq, m, 32); }
  float mean = s * (1.f/64.f);
  float var  = sq * (1.f/64.f) - mean*mean;
  float inv  = rsqrtf(var + 1e-5f);
  pair[(size_t)wid*PCH + lane]      = (acc0 - mean)*inv*pln_s[lane]      + pln_o[lane];
  pair[(size_t)wid*PCH + lane + 32] = (acc1 - mean)*inv*pln_s[lane + 32] + pln_o[lane + 32];
}

// ---------------------------------------------------------------------------
// Attention: one wave per (n, head). Lane = neighbor k for logits/softmax,
// lane = output dim d for the weighted-V accumulation. q row staged to LDS
// via async global->LDS copy (gfx1250 GLOBAL_LOAD_ASYNC_TO_LDS_B32).
// ---------------------------------------------------------------------------
__global__ __launch_bounds__(256) void attn_kernel(
    const float* __restrict__ q, const float* __restrict__ kbuf,
    const float* __restrict__ vbuf, const float* __restrict__ pair,
    const int* __restrict__ pm, const int* __restrict__ nb,
    const float* __restrict__ wb, float* __restrict__ o)
{
  __shared__ float qs[LCH];
  const int n = blockIdx.x, tid = threadIdx.x;
#if ASYNC_LDS
  __builtin_amdgcn_global_load_async_to_lds_b32(
      (as1_int*)(q + (size_t)n*LCH + tid),
      (as3_int*)&qs[tid], 0, 0);
  __builtin_amdgcn_s_wait_asynccnt(0);
  __syncthreads();
#else
  qs[tid] = q[(size_t)n*LCH + tid];
  __syncthreads();
#endif
  const int h = tid >> 5, lane = tid & 31;

  const int idx = nb[(size_t)n*KNB + lane];
  const int i = idx < 0 ? 0 : idx;

  float logit = 0.f;
  const float* kp = kbuf + (size_t)i*LCH + h*32;
  #pragma unroll
  for (int d = 0; d < 32; ++d) logit += qs[h*32 + d] * kp[d];
  logit *= 0.17677669529663687f;   // 1/sqrt(32)

  const float* pp = pair + (size_t)(n*KNB + lane) * PCH;
  float bsum = 0.f;
  #pragma unroll
  for (int p = 0; p < PCH; ++p) bsum += pp[p] * wb[p*HEADS + h];
  logit += bsum;
  if (!pm[(size_t)n*KNB + lane]) logit = -1e9f;

  float mx = logit;
  for (int m = 16; m; m >>= 1) mx = fmaxf(mx, __shfl_xor(mx, m, 32));
  float e = __expf(logit - mx);
  float se = e;
  for (int m = 16; m; m >>= 1) se += __shfl_xor(se, m, 32);
  float attn = e / se;

  float acc = 0.f;
  for (int kk = 0; kk < 32; ++kk) {
    float a = __shfl(attn, kk, 32);
    int  ii = __shfl(i,    kk, 32);
    acc += a * vbuf[(size_t)ii*LCH + h*32 + lane];
  }
  o[(size_t)n*LCH + h*32 + lane] = acc;
}

// ---------------------------------------------------------------------------
// LayerNorm over 256: one wave per row. mode 0: out = LN(x + res);
// mode 1: out = res + LN(x).
// ---------------------------------------------------------------------------
__global__ __launch_bounds__(256) void ln_kernel(
    float* __restrict__ out, const float* __restrict__ x,
    const float* __restrict__ res, const float* __restrict__ s,
    const float* __restrict__ o, int M, int mode)
{
  const int row = blockIdx.x * 8 + (threadIdx.x >> 5);
  if (row >= M) return;
  const int lane = threadIdx.x & 31;
  float v[8], rv[8];
  #pragma unroll
  for (int j = 0; j < 8; ++j) {
    int col = j*32 + lane;
    v[j]  = x[(size_t)row*LCH + col];
    rv[j] = res ? res[(size_t)row*LCH + col] : 0.f;
    if (mode == 0) v[j] += rv[j];
  }
  float sum = 0.f, sq = 0.f;
  #pragma unroll
  for (int j = 0; j < 8; ++j) { sum += v[j]; sq += v[j]*v[j]; }
  for (int m = 16; m; m >>= 1) { sum += __shfl_xor(sum, m, 32); sq += __shfl_xor(sq, m, 32); }
  float mean = sum * (1.f/256.f);
  float var  = sq  * (1.f/256.f) - mean*mean;
  float inv  = rsqrtf(var + 1e-5f);
  #pragma unroll
  for (int j = 0; j < 8; ++j) {
    int col = j*32 + lane;
    float y = (v[j] - mean) * inv * s[col] + o[col];
    if (mode == 1) y += rv[j];
    out[(size_t)row*LCH + col] = y;
  }
}

__global__ void add_kernel(float* __restrict__ y, const float* __restrict__ x, int n) {
  int i = blockIdx.x * 256 + threadIdx.x;
  if (i < n) y[i] += x[i];
}
__global__ void mul_kernel(float* __restrict__ d, const float* __restrict__ a,
                           const float* __restrict__ b, int n) {
  int i = blockIdx.x * 256 + threadIdx.x;
  if (i < n) d[i] = a[i] * b[i];
}
__global__ void copy_kernel(float* __restrict__ d, const float* __restrict__ s, int n) {
  int i = blockIdx.x * 256 + threadIdx.x;
  if (i < n) d[i] = s[i];
}

// ---------------------------------------------------------------------------
// Final: augment = mean_norm(loc2[:,5:] + upd); pos_out = R @ loc_new + t.
// ---------------------------------------------------------------------------
__global__ __launch_bounds__(256) void final_assemble_kernel(
    const float* __restrict__ loc2, const float* __restrict__ upd,
    const float* __restrict__ Rb, const float* __restrict__ Tb,
    float* __restrict__ pos_out)
{
  int n = blockIdx.x * blockDim.x + threadIdx.x;
  if (n >= NRES) return;
  float lnew[21][3]; float ms = 0.f;
  for (int a = 0; a < 5; ++a)
    for (int d = 0; d < 3; ++d) lnew[a][d] = loc2[(size_t)n*63 + a*3 + d];
  for (int j = 0; j < 16; ++j)
    for (int d = 0; d < 3; ++d) {
      float a = loc2[(size_t)n*63 + (5+j)*3 + d] + upd[(size_t)n*48 + j*3 + d];
      lnew[5+j][d] = a; ms += a*a;
    }
  ms = sqrtf(ms * (1.f/16.f) + FEPS);
  for (int j = 0; j < 16; ++j)
    for (int d = 0; d < 3; ++d) lnew[5+j][d] /= ms;

  float Rm[9], t[3];
  for (int j = 0; j < 9; ++j) Rm[j] = Rb[(size_t)n*9 + j];
  for (int d = 0; d < 3; ++d) t[d] = Tb[(size_t)n*3 + d];
  for (int a = 0; a < 21; ++a)
    for (int j = 0; j < 3; ++j)
      pos_out[(size_t)n*63 + a*3 + j] =
        Rm[j*3+0]*lnew[a][0] + Rm[j*3+1]*lnew[a][1] + Rm[j*3+2]*lnew[a][2] + t[j];
}

// ---------------------------------------------------------------------------
// Block-parameter index map (JAX pytree flattens dicts in sorted-key order).
// ---------------------------------------------------------------------------
enum { I_BO=0, I_LN1_O, I_LN1_S, I_LN2_O, I_LN2_S, I_PLN_O, I_PLN_S,
       I_PM_B1, I_PM_B2, I_PM_W1, I_PM_W2,
       I_UP_BO, I_UP_PB1, I_UP_PB2, I_UP_PW1, I_UP_PW2, I_UP_WG, I_UP_WO, I_UP_WU,
       I_W_DIR, I_W_DIST, I_W_REL, I_W_ROT, I_W_VEC,
       I_WB, I_WK, I_WO, I_WQ, I_WV };

extern "C" void kernel_launch(void* const* d_in, const int* in_sizes, int n_in,
                              void* d_out, int out_size, void* d_ws, size_t ws_size,
                              hipStream_t stream)
{
  const float*         pos   = (const float*)d_in[0];
  const int*           resi  = (const int*)d_in[1];
  const int*           chain = (const int*)d_in[2];
  const int*           batch = (const int*)d_in[3];
  const unsigned char* mask  = (const unsigned char*)d_in[4];
  const unsigned char* is_t  = (const unsigned char*)d_in[5];
  const float*         hot   = (const float*)d_in[6];
  const int*           aa    = (const int*)d_in[7];
  auto BPf = [&](int b, int i) -> const float* {
    return (const float*)d_in[8 + 29*b + i];
  };
  const float* f_b1  = (const float*)d_in[66];
  const float* f_b2  = (const float*)d_in[67];
  const float* f_w1  = (const float*)d_in[68];
  const float* f_w2  = (const float*)d_in[69];
  const float* lnf_o = (const float*)d_in[70];
  const float* lnf_s = (const float*)d_in[71];
  const float* p_ln_o  = (const float*)d_in[72];
  const float* p_ln_s  = (const float*)d_in[73];
  const float* p_mlp_w1= (const float*)d_in[74];
  const float* p_mlp_w2= (const float*)d_in[75];
  const float* p_vec_w = (const float*)d_in[76];

  // workspace layout
  char* W = (char*)d_ws; size_t off = 0;
  auto WA = [&](size_t bytes) -> void* {
    void* p = W + off; off += (bytes + 255) & ~(size_t)255; return p;
  };
  float* Rb    = (float*)WA((size_t)NRES*9*4);
  float* Tb    = (float*)WA((size_t)NRES*3*4);
  float* loc2  = (float*)WA((size_t)NRES*63*4);
  float* pos21 = (float*)WA((size_t)NRES*63*4);
  float* feats = (float*)WA((size_t)NRES*448*4);
  int*   nb    = (int*)  WA((size_t)NRES*KNB*4);
  int*   pm    = (int*)  WA((size_t)NK*4);
  float* x     = (float*)WA((size_t)NRES*LCH*4);
  float* inc   = (float*)WA((size_t)NRES*LCH*4);
  float* tmp1  = (float*)WA((size_t)NRES*1024*4);
  float* qb    = (float*)WA((size_t)NRES*LCH*4);
  float* kb    = (float*)WA((size_t)NRES*LCH*4);
  float* vb    = (float*)WA((size_t)NRES*LCH*4);
  float* attno = (float*)WA((size_t)NRES*LCH*4);
  float* upd   = (float*)WA((size_t)NRES*LCH*4);
  float* x2    = (float*)WA((size_t)NRES*LCH*4);
  float* pair  = (float*)WA((size_t)NK*PCH*4);
  float* pairh = (float*)WA((size_t)65536*128*4);  // pair-MLP hidden chunk / gate reuse

  auto gemm = [&](const float* A, int lda, const float* B, int ldb,
                  const float* bias, const float* res, float* C, int ldc,
                  int M, int Kd, int Nn, int flags) {
    dim3 g((M + 63) / 64, (Nn + 63) / 64);
    wmma_gemm_kernel<<<g, 256, 0, stream>>>(A, lda, B, ldb, bias, res, C, ldc,
                                            M, Kd, Nn, flags);
  };

  // ---- prepare ----
  prep_kernel<<<NRES/256, 256, 0, stream>>>(pos, is_t, hot, aa, p_vec_w,
                                            Rb, Tb, loc2, pos21, feats);
  knn_kernel<<<NRES, 256, 0, stream>>>(pos21, batch, is_t, mask, nb);

  gemm(feats, 448, p_mlp_w1, 1024, nullptr, nullptr, tmp1, 1024,
       NRES, 443, 1024, FG_GELU);
  gemm(tmp1, 1024, p_mlp_w2, 256, nullptr, nullptr, x, 256,
       NRES, 1024, 256, 0);
  ln_kernel<<<NRES/8, 256, 0, stream>>>(x, x, nullptr, p_ln_s, p_ln_o, NRES, 0);
  copy_kernel<<<NRES*LCH/256, 256, 0, stream>>>(inc, x, NRES*LCH);

  // ---- blocks ----
  for (int b = 0; b < 2; ++b) {
    pair_kernel<<<NK/8, 256, 0, stream>>>(pos21, Rb, Tb, nb, resi, chain, mask,
        BPf(b, I_W_REL), BPf(b, I_W_DIST), BPf(b, I_W_DIR), BPf(b, I_W_VEC),
        BPf(b, I_W_ROT), BPf(b, I_PLN_S), BPf(b, I_PLN_O), pair, pm);

    for (int c = 0; c < 4; ++c) {           // pair MLP in 64K-row chunks
      float* pr = pair + (size_t)c * 65536 * PCH;
      gemm(pr, PCH, BPf(b, I_PM_W1), 128, BPf(b, I_PM_B1), nullptr,
           pairh, 128, 65536, PCH, 128, FG_GELU | FG_BIAS);
      gemm(pairh, 128, BPf(b, I_PM_W2), PCH, BPf(b, I_PM_B2), nullptr,
           pr, PCH, 65536, 128, PCH, FG_BIAS);
    }

    gemm(x, 256, BPf(b, I_WQ), 256, nullptr, nullptr, qb, 256, NRES, 256, 256, 0);
    gemm(x, 256, BPf(b, I_WK), 256, nullptr, nullptr, kb, 256, NRES, 256, 256, 0);
    gemm(x, 256, BPf(b, I_WV), 256, nullptr, nullptr, vb, 256, NRES, 256, 256, 0);
    attn_kernel<<<NRES, 256, 0, stream>>>(qb, kb, vb, pair, pm, nb,
                                          BPf(b, I_WB), attno);
    gemm(attno, 256, BPf(b, I_WO), 256, BPf(b, I_BO), nullptr, upd, 256,
         NRES, 256, 256, FG_BIAS);
    add_kernel<<<NRES*LCH/256, 256, 0, stream>>>(inc, upd, NRES*LCH);
    ln_kernel<<<NRES/8, 256, 0, stream>>>(x, x, upd, BPf(b, I_LN1_S),
                                          BPf(b, I_LN1_O), NRES, 0);

    // update: x2 = x + posMLP(loc2); upd = (gelu(x2 Wg) * (x2 Wu)) Wo + bo
    gemm(loc2, 63, BPf(b, I_UP_PW1), 512, BPf(b, I_UP_PB1), nullptr,
         tmp1, 512, NRES, 63, 512, FG_GELU | FG_BIAS);
    gemm(tmp1, 512, BPf(b, I_UP_PW2), 256, BPf(b, I_UP_PB2), x,
         x2, 256, NRES, 512, 256, FG_BIAS | FG_RES);
    gemm(x2, 256, BPf(b, I_UP_WU), 512, nullptr, nullptr, tmp1, 512,
         NRES, 256, 512, 0);                                  // lu
    gemm(x2, 256, BPf(b, I_UP_WG), 512, nullptr, nullptr,
         tmp1 + (size_t)NRES*512, 512, NRES, 256, 512, FG_GELU);  // lg
    mul_kernel<<<NRES*512/256, 256, 0, stream>>>(pairh, tmp1,
        tmp1 + (size_t)NRES*512, NRES*512);
    gemm(pairh, 512, BPf(b, I_UP_WO), 256, BPf(b, I_UP_BO), nullptr,
         upd, 256, NRES, 512, 256, FG_BIAS);
    add_kernel<<<NRES*LCH/256, 256, 0, stream>>>(inc, upd, NRES*LCH);
    ln_kernel<<<NRES/8, 256, 0, stream>>>(x, x, upd, BPf(b, I_LN2_S),
                                          BPf(b, I_LN2_O), NRES, 0);
  }

  // ---- final ----
  ln_kernel<<<NRES/8, 256, 0, stream>>>(x, inc, x, lnf_s, lnf_o, NRES, 1);
  copy_kernel<<<NRES*LCH/256, 256, 0, stream>>>((float*)d_out, x, NRES*LCH);
  gemm(x, 256, f_w1, 512, f_b1, nullptr, tmp1, 512, NRES, 256, 512,
       FG_GELU | FG_BIAS);
  gemm(tmp1, 512, f_w2, 48, f_b2, nullptr, upd, 48, NRES, 512, 48, FG_BIAS);
  final_assemble_kernel<<<NRES/256, 256, 0, stream>>>(loc2, upd, Rb, Tb,
      (float*)d_out + (size_t)NRES*LCH);
}
